// GNN_438086664596
// MI455X (gfx1250) — matrix operand order repaired and checked by reading, not compiled, not run on hardware
//
#include <hip/hip_runtime.h>
#include <math.h>

// ---------------------------------------------------------------------------
// GATv2 GNN forward for MI455X (gfx1250, wave32).
//   N=20000 nodes, E=640000 edges, H=2 heads, D=64, L=4 layers.
// GEMMs: v_wmma_f32_16x16x32_f16 (f16 in / f32 accum), A-tile staged into LDS
// via the Tensor Data Mover (tensor_load_to_lds + s_wait_tensorcnt), with TDM
// LDS padding producing the 80-half pitch needed for aligned fragment loads.
// Edge phase: wave-per-edge gather/scatter through the 192MB L2 (xl/xr/agg
// ~10MB each, fully L2 resident), f32 atomics for segment max/sum.
// ---------------------------------------------------------------------------

typedef __attribute__((ext_vector_type(16))) _Float16 v16h;
typedef __attribute__((ext_vector_type(8)))  _Float16 v8h;
typedef __attribute__((ext_vector_type(8)))  float    v8f;
typedef __attribute__((ext_vector_type(4)))  unsigned int u32x4;
typedef __attribute__((ext_vector_type(8)))  int      i32x8;
typedef __attribute__((ext_vector_type(4)))  int      i32x4;

#define DHID 64
#define HEADS 2
#define TILE_ROWS 64
#define LDS_PITCH 80   // halves; 160B rows -> 32B-aligned vector LDS loads

#if __has_builtin(__builtin_amdgcn_tensor_load_to_lds)
  #define HAVE_TDM 1
  #if __has_include(<hip/amd_detail/amd_gfx1250_TDM.h>)
    // amdgpu-toolchain (clang-23 / therock): 6-arg form
    #define TDM_LOAD(g0, g1, g2, g3) \
        __builtin_amdgcn_tensor_load_to_lds((g0), (g1), (g2), (g3), \
            (i32x8){0,0,0,0,0,0,0,0}, 0)
  #else
    // ROCm 7.2 (clang-22): 5-arg form
    #define TDM_LOAD(g0, g1, g2, g3) \
        __builtin_amdgcn_tensor_load_to_lds((g0), (g1), (g2), (g3), 0)
  #endif
#else
  #define HAVE_TDM 0
#endif

__device__ __forceinline__ void atomicMaxF(float* addr, float v) {
    // signed-int max for non-negative floats, unsigned min for negative floats
    if (v >= 0.0f) atomicMax((int*)addr, __float_as_int(v));
    else           atomicMin((unsigned int*)addr, (unsigned int)__float_as_int(v));
}

// ---------------------------------------------------------------------------
// GEMM: out[M x NCOLS] = Xh[M x 64](f16) * W[64 x NCOLS](f32->f16) + bias.
// One block = 64 rows x NCOLS, 256 threads = 8 wave32s.
// A tile comes in via TDM (async tensor DMA) with pad_interval=32dw,
// pad_amount=8dw -> LDS row pitch of 40 dwords = 80 halves = LDS_PITCH.
// ---------------------------------------------------------------------------
template<int NCOLS, bool ACT>
__global__ __launch_bounds__(256)
void gemm_k64_wmma(const _Float16* __restrict__ Xh, const float* __restrict__ W,
                   const float* __restrict__ bias, float* __restrict__ out,
                   int M, float slope) {
    __shared__ _Float16 Abuf[TILE_ROWS * LDS_PITCH];
    __shared__ _Float16 Wt[NCOLS * LDS_PITCH];

    const int tid = threadIdx.x;
    const int rowBase = blockIdx.x * TILE_ROWS;
    const int rows = min(TILE_ROWS, M - rowBase);

#if HAVE_TDM
    if (tid < 32) {
        // Tensor DMA descriptor (D#): 2D f16 tensor, tile 64x64, zero-fill
        // rows past tensor_dim1, LDS padding 8 dwords every 32 dwords.
        unsigned lds = (unsigned)(size_t)(&Abuf[0]);
        unsigned long long ga =
            (unsigned long long)(size_t)(Xh + (size_t)rowBase * DHID);
        u32x4 g0 = { 1u,                                  // count=1, user mode
                     lds,                                 // lds_addr
                     (unsigned)(ga & 0xffffffffu),        // global_addr[31:0]
                     (unsigned)((ga >> 32) & 0x1ffffffu)  // global_addr[56:32]
                         | (2u << 30) };                  // type=2 (image)
        i32x8 g1 = { (int)((1u << 16)      // data_size = 1 -> 2 bytes
                         | (1u << 20)      // pad_enable
                         | (4u << 22)      // pad_interval: 32 dwords (128B row)
                         | (7u << 25)),    // pad_amount:   8 dwords (16 halves)
                     (int)(64u << 16),                  // tensor_dim0 = 64
                     (int)(((unsigned)rows) << 16),     // tensor_dim1 = rows
                     (int)(64u << 16),                  // tile_dim0 = 64
                     64,                                // tile_dim1 = 64
                     64,                                // tensor_dim0_stride
                     0, 0 };
        i32x4 gz = { 0, 0, 0, 0 };
        TDM_LOAD(g0, g1, gz, gz);
        __builtin_amdgcn_s_wait_tensorcnt(0);
    }
#else
    for (int i = tid; i < TILE_ROWS * DHID; i += 256) {
        int r = i >> 6, c = i & 63;
        _Float16 v = (_Float16)0.0f;
        if (r < rows) v = Xh[(size_t)(rowBase + r) * DHID + c];
        Abuf[r * LDS_PITCH + c] = v;
    }
#endif
    // W transposed into LDS: Wt[n][k] = (f16)W[k][n]
    for (int i = tid; i < DHID * NCOLS; i += 256) {
        int k = i / NCOLS, n = i % NCOLS;
        Wt[n * LDS_PITCH + k] = (_Float16)W[i];
    }
    __syncthreads();

    const int wave = tid >> 5, lane = tid & 31;
    const int laneLo = lane & 15, laneHi = lane >> 4;
    const int nColTiles = NCOLS >> 4;
    const int nTiles = (TILE_ROWS / 16) * nColTiles;
    const bool fullTile = (rows == TILE_ROWS);

    for (int t = wave; t < nTiles; t += 8) {
        const int ct = t % nColTiles;
        const int rt = t / nColTiles;
        const int n = (ct << 4) + laneLo;          // output column of this lane
        const int arow = (rt << 4) + laneLo;       // A row within tile

        v8f c;
        const float bv = bias[n];
        #pragma unroll
        for (int r = 0; r < 8; ++r) c[r] = bv;

        #pragma unroll
        for (int k0 = 0; k0 < DHID; k0 += 32) {
            // A fragment (16x32 f16): lanes 0-15 hold K {0..7,16..23},
            // lanes 16-31 hold K {8..15,24..31} of their row.
            union { v16h v; v8h h[2]; } af;
            const _Float16* ap = &Abuf[arow * LDS_PITCH + k0 + (laneHi << 3)];
            af.h[0] = *(const v8h*)ap;
            af.h[1] = *(const v8h*)(ap + 16);
            // B fragment (32x16 f16): lanes 0-15 hold K {0..7,16..23} of column n,
            // lanes 16-31 hold K {8..15,24..31}.
            union { v16h v; v8h h[2]; } bfu;
            const _Float16* bp = &Wt[n * LDS_PITCH + k0 + (laneHi << 3)];
            bfu.h[0] = *(const v8h*)bp;
            bfu.h[1] = *(const v8h*)(bp + 16);
            v16h bf = bfu.v;
            c = __builtin_amdgcn_wmma_f32_16x16x32_f16(
                    false, af.v, false, bf, (short)0, c, false, false);
        }

        // C/D layout: VGPR r -> row r (lanes 0-15) / row 8+r (lanes 16-31)
        const int rstore = rowBase + (rt << 4) + (laneHi << 3);
        if (fullTile) {
            #pragma unroll
            for (int r = 0; r < 8; ++r) {
                float v = c[r];
                if (ACT) v = v > 0.0f ? v : slope * v;
                out[(size_t)(rstore + r) * NCOLS + n] = v;
            }
        } else {
            #pragma unroll
            for (int r = 0; r < 8; ++r) {
                int row = rstore + r;
                if (row < M) {
                    float v = c[r];
                    if (ACT) v = v > 0.0f ? v : slope * v;
                    out[(size_t)row * NCOLS + n] = v;
                }
            }
        }
    }
}

// ---------------------------------------------------------------------------
__global__ __launch_bounds__(256)
void to_half(const float* __restrict__ x, _Float16* __restrict__ xh, int count) {
    int i = blockIdx.x * 256 + threadIdx.x;
    if (i < count) xh[i] = (_Float16)x[i];
}

// ---------------------------------------------------------------------------
// Per-layer scratch init: agg=0, m=-inf, denom=0
// ---------------------------------------------------------------------------
__global__ __launch_bounds__(256)
void init_layer(float* __restrict__ agg, float* __restrict__ m,
                float* __restrict__ denom, int N) {
    int i = blockIdx.x * 256 + threadIdx.x;
    if (i < N * HEADS * DHID) agg[i] = 0.0f;
    if (i < N * HEADS) { m[i] = -INFINITY; denom[i] = 0.0f; }
}

// ---------------------------------------------------------------------------
// Wave-per-edge: logits[e][h] = sum_d att[h][d]*lrelu(xl[src][h][d]+xr[dst][h][d])
// plus segment-max into m[dst][h].
// ---------------------------------------------------------------------------
__global__ __launch_bounds__(256)
void edge_logits(const int* __restrict__ src, const int* __restrict__ dst,
                 const float* __restrict__ xl, const float* __restrict__ xr,
                 const float* __restrict__ att, float* __restrict__ logits,
                 float* __restrict__ m, int E) {
    const int gtid = blockIdx.x * 256 + threadIdx.x;
    const int wid = gtid >> 5, lane = gtid & 31;
    const int nWaves = (gridDim.x * 256) >> 5;

    // att is reused by every edge: hoist this lane's 4 coefficients
    const float at0 = att[lane],      at1 = att[lane + 32];
    const float at2 = att[lane + 64], at3 = att[lane + 96];

    for (int e = wid; e < E; e += nWaves) {
        const int s = src[e], d = dst[e];
        const float* pl = xl + (size_t)s * (HEADS * DHID);
        const float* pr = xr + (size_t)d * (HEADS * DHID);

        float v0 = pl[lane]      + pr[lane];
        float v1 = pl[lane + 32] + pr[lane + 32];
        float v2 = pl[lane + 64] + pr[lane + 64];
        float v3 = pl[lane + 96] + pr[lane + 96];
        v0 = v0 > 0.0f ? v0 : 0.2f * v0;
        v1 = v1 > 0.0f ? v1 : 0.2f * v1;
        v2 = v2 > 0.0f ? v2 : 0.2f * v2;
        v3 = v3 > 0.0f ? v3 : 0.2f * v3;
        float a0 = v0 * at0 + v1 * at1;   // head 0
        float a1 = v2 * at2 + v3 * at3;   // head 1

        #pragma unroll
        for (int off = 16; off > 0; off >>= 1) {
            a0 += __shfl_down(a0, off, 32);
            a1 += __shfl_down(a1, off, 32);
        }
        if (lane == 0) {
            logits[2 * e]     = a0;
            logits[2 * e + 1] = a1;
            atomicMaxF(&m[2 * d],     a0);
            atomicMaxF(&m[2 * d + 1], a1);
        }
    }
}

// ---------------------------------------------------------------------------
// ex = exp(logits - m[dst]); denom[dst] += ex  (thread per (edge,head))
// ---------------------------------------------------------------------------
__global__ __launch_bounds__(256)
void edge_exp_denom(const int* __restrict__ dst, const float* __restrict__ logits,
                    const float* __restrict__ m, float* __restrict__ ex,
                    float* __restrict__ denom, int E) {
    int i = blockIdx.x * 256 + threadIdx.x;
    if (i >= E * HEADS) return;
    int e = i >> 1, h = i & 1;
    int d = dst[e];
    float v = __expf(logits[i] - m[2 * d + h]);
    ex[i] = v;
    atomicAdd(&denom[2 * d + h], v);
}

// ---------------------------------------------------------------------------
// Wave-per-edge scatter: agg[dst] += xl[src] * alpha  (128 floats per edge)
// ---------------------------------------------------------------------------
__global__ __launch_bounds__(256)
void edge_scatter(const int* __restrict__ src, const int* __restrict__ dst,
                  const float* __restrict__ xl, const float* __restrict__ ex,
                  const float* __restrict__ denom, float* __restrict__ agg, int E) {
    const int gtid = blockIdx.x * 256 + threadIdx.x;
    const int wid = gtid >> 5, lane = gtid & 31;
    const int nWaves = (gridDim.x * 256) >> 5;

    for (int e = wid; e < E; e += nWaves) {
        const int s = src[e], d = dst[e];
        const float al0 = ex[2 * e]     / (denom[2 * d]     + 1e-16f);
        const float al1 = ex[2 * e + 1] / (denom[2 * d + 1] + 1e-16f);
        const float* pl = xl + (size_t)s * (HEADS * DHID);
        float* pa = agg + (size_t)d * (HEADS * DHID);
        #pragma unroll
        for (int k = 0; k < 4; ++k) {
            int j = lane + (k << 5);
            float a = (j < DHID) ? al0 : al1;
            atomicAdd(&pa[j], pl[j] * a);
        }
    }
}

// ---------------------------------------------------------------------------
// x_next[n][d] = lrelu(0.5*(agg[n][0][d]+agg[n][1][d]) + bias[d], 0.01)
// Written directly as f16 (the WMMA input precision for the next layer).
// ---------------------------------------------------------------------------
__global__ __launch_bounds__(256)
void node_finalize(const float* __restrict__ agg, const float* __restrict__ bias,
                   _Float16* __restrict__ xnext, int N) {
    int i = blockIdx.x * 256 + threadIdx.x;
    if (i >= N * DHID) return;
    int n = i >> 6, d = i & 63;
    float v = 0.5f * (agg[(size_t)n * 128 + d] + agg[(size_t)n * 128 + 64 + d]) + bias[d];
    v = v > 0.0f ? v : 0.01f * v;
    xnext[i] = (_Float16)v;
}

// ---------------------------------------------------------------------------
extern "C" void kernel_launch(void* const* d_in, const int* in_sizes, int n_in,
                              void* d_out, int out_size, void* d_ws, size_t ws_size,
                              hipStream_t stream) {
    (void)n_in; (void)ws_size; (void)out_size;
    const int*   edge_index = (const int*)  d_in[0];   // [2,E]
    const float* pert_emb   = (const float*)d_in[2];   // [N,64]
    const float* Wl         = (const float*)d_in[3];   // [L,64,128]
    const float* bl         = (const float*)d_in[4];   // [L,128]
    const float* Wr         = (const float*)d_in[5];
    const float* br         = (const float*)d_in[6];
    const float* att        = (const float*)d_in[7];   // [L,2,64]
    const float* bias       = (const float*)d_in[8];   // [L,64]
    const float* Wo         = (const float*)d_in[9];   // [64,64]
    const float* bo         = (const float*)d_in[10];  // [64]

    const int E = in_sizes[0] / 2;
    const int N = in_sizes[2] / DHID;
    const int L = in_sizes[8] / DHID;

    const int* src = edge_index;
    const int* dst = edge_index + E;

    // workspace carve-up (all offsets 256B aligned)
    char* ws = (char*)d_ws;
    size_t off = 0;
    float*    xl    = (float*)(ws + off); off += (size_t)N * 128 * 4;  // 10.24 MB
    float*    xr    = (float*)(ws + off); off += (size_t)N * 128 * 4;
    float*    agg   = (float*)(ws + off); off += (size_t)N * 128 * 4;
    _Float16* xh    = (_Float16*)(ws + off); off += (size_t)N * 64 * 2;
    float*    logit = (float*)(ws + off); off += (size_t)E * 2 * 4;
    float*    exb   = (float*)(ws + off); off += (size_t)E * 2 * 4;
    float*    mbuf  = (float*)(ws + off); off += (size_t)N * 2 * 4;
    float*    denom = (float*)(ws + off); off += (size_t)N * 2 * 4;

    const int gemmBlocks  = (N + TILE_ROWS - 1) / TILE_ROWS;
    const int cvtBlocks   = (N * 64 + 255) / 256;
    const int initBlocks  = (N * 128 + 255) / 256;
    const int waveBlocks  = (E + 7) / 8;           // 8 wave32s per block
    const int ehBlocks    = (E * 2 + 255) / 256;
    const int finBlocks   = (N * 64 + 255) / 256;

    to_half<<<cvtBlocks, 256, 0, stream>>>(pert_emb, xh, N * DHID);

    for (int l = 0; l < L; ++l) {
        const float* Wl_l = Wl + (size_t)l * DHID * 128;
        const float* bl_l = bl + (size_t)l * 128;
        const float* Wr_l = Wr + (size_t)l * DHID * 128;
        const float* br_l = br + (size_t)l * 128;
        const float* at_l = att + (size_t)l * 128;
        const float* bi_l = bias + (size_t)l * DHID;

        gemm_k64_wmma<128, false><<<gemmBlocks, 256, 0, stream>>>(xh, Wl_l, bl_l, xl, N, 0.0f);
        gemm_k64_wmma<128, false><<<gemmBlocks, 256, 0, stream>>>(xh, Wr_l, br_l, xr, N, 0.0f);
        init_layer<<<initBlocks, 256, 0, stream>>>(agg, mbuf, denom, N);
        edge_logits<<<waveBlocks, 256, 0, stream>>>(src, dst, xl, xr, at_l, logit, mbuf, E);
        edge_exp_denom<<<ehBlocks, 256, 0, stream>>>(dst, logit, mbuf, exb, denom, E);
        edge_scatter<<<waveBlocks, 256, 0, stream>>>(src, dst, xl, exb, denom, agg, E);
        node_finalize<<<finBlocks, 256, 0, stream>>>(agg, bi_l, xh, N);
    }

    // final projection + leaky_relu(0.01) straight into d_out
    gemm_k64_wmma<64, true><<<gemmBlocks, 256, 0, stream>>>(xh, Wo, bo, (float*)d_out, N, 0.01f);
}
